// GLMAttention_67396626809289
// MI455X (gfx1250) — matrix operand order repaired and checked by reading, not compiled, not run on hardware
//
#include <hip/hip_runtime.h>
#include <hip/hip_bf16.h>

// ---------------------------------------------------------------- constants
#define BATCH 2
#define SEQ   2048
#define HID   4096
#define NHEAD 32
#define NKVH  2
#define HDIM  128
#define ROTD  64
#define QKVD  ((NHEAD + 2 * NKVH) * HDIM)   // 4608
#define MROWS (BATCH * SEQ)                  // 4096

typedef __attribute__((ext_vector_type(16))) __bf16 bf16x16;
typedef __attribute__((ext_vector_type(8)))  __bf16 bf16x8;
typedef __attribute__((ext_vector_type(8)))  float  f32x8;

union Frag16 {
    bf16x16 v;
    bf16x8  h[2];
};

// -------- gfx1250 async global->LDS path (guarded; sync fallback otherwise)
#if defined(__has_builtin)
#if __has_builtin(__builtin_amdgcn_global_load_async_to_lds_b128)
#define HAVE_ASYNC_LDS 1
#endif
#if __has_builtin(__builtin_amdgcn_s_wait_asynccnt)
#define WAIT_ASYNC(n) __builtin_amdgcn_s_wait_asynccnt(n)
#endif
#endif
#if defined(HAVE_ASYNC_LDS) && !defined(WAIT_ASYNC)
#define WAIT_ASYNC(n) asm volatile("s_wait_asynccnt %0" ::"n"(n) : "memory")
#endif

#ifdef HAVE_ASYNC_LDS
typedef int v4i_t __attribute__((vector_size(16)));
__device__ __forceinline__ void async_copy_b128(const __bf16* g, __bf16* l) {
    // builtin takes v4i pointers: global (AS1) source, LDS (AS3) destination
    __builtin_amdgcn_global_load_async_to_lds_b128(
        (__attribute__((address_space(1))) v4i_t*)(v4i_t*)(g),
        (__attribute__((address_space(3))) v4i_t*)(v4i_t*)(l), 0, 0);
}
#endif

// ============================================================== convert f32->bf16
__global__ __launch_bounds__(256)
void cvt_f32_bf16(const float* __restrict__ src, __bf16* __restrict__ dst, size_t n) {
    size_t i = ((size_t)blockIdx.x * blockDim.x + threadIdx.x) * 4;
    size_t stride = (size_t)gridDim.x * blockDim.x * 4;
    for (; i < n; i += stride) {
        float4 f = *(const float4*)(src + i);
        dst[i + 0] = (__bf16)f.x;
        dst[i + 1] = (__bf16)f.y;
        dst[i + 2] = (__bf16)f.z;
        dst[i + 3] = (__bf16)f.w;
    }
}

// ============================================================== convert+transpose
// src f32 [K][N]  ->  dst bf16 [N][K]   (one-time weight prep; LDS-tiled)
__global__ __launch_bounds__(256)
void cvt_transpose_bf16(const float* __restrict__ src, __bf16* __restrict__ dst,
                        int K, int N) {
    __shared__ __bf16 t[64][64 + 8];
    const int k0 = blockIdx.y * 64, n0 = blockIdx.x * 64;
    {
        int kr = threadIdx.x >> 2;           // 0..63 source row
        int nc = (threadIdx.x & 3) * 16;     // 16 source cols
        const float* s = src + (size_t)(k0 + kr) * N + n0 + nc;
#pragma unroll
        for (int i = 0; i < 16; i += 4) {
            float4 f = *(const float4*)(s + i);
            t[nc + i + 0][kr] = (__bf16)f.x;
            t[nc + i + 1][kr] = (__bf16)f.y;
            t[nc + i + 2][kr] = (__bf16)f.z;
            t[nc + i + 3][kr] = (__bf16)f.w;
        }
    }
    __syncthreads();
    {
        int nr = threadIdx.x >> 2;           // 0..63 dst row (n)
        int kc = (threadIdx.x & 3) * 16;     // 16 dst cols (k)
        __bf16* d = dst + (size_t)(n0 + nr) * K + k0 + kc;
        *(bf16x8*)(d)     = *(const bf16x8*)&t[nr][kc];
        *(bf16x8*)(d + 8) = *(const bf16x8*)&t[nr][kc + 8];
    }
}

// ============================================================== bf16 WMMA GEMM
// C[M,N](f32) = A[M,K](bf16 row-major) * Bt[N,K](bf16, pre-transposed) + bias
// Block tile 128x128, BK=32, 128 threads = 4 waves in 2x2, wave tile 64x64
// (4x4 WMMA accums -> 16 wmma per 16 ds_b128 fragment loads per K-step).
// Double-buffered LDS staged with GLOBAL_LOAD_ASYNC_TO_LDS_B128 when available.
#define GTM 128
#define GTN 128
#define GTK 32
#define KPAD 8   // row stride 40 elems = 80B (16B-aligned)

__global__ __launch_bounds__(128)
void gemm_bf16(const __bf16* __restrict__ A, const __bf16* __restrict__ Bt,
               const float* __restrict__ bias, float* __restrict__ C,
               int M, int N, int K) {
    __shared__ __bf16 lA[2][GTM][GTK + KPAD];
    __shared__ __bf16 lB[2][GTN][GTK + KPAD];

    const int tid  = threadIdx.x;
    const int lane = tid & 31;
    const int wave = tid >> 5;           // 0..3
    const int wr   = (wave >> 1) * 64;   // wave M offset
    const int wc   = (wave & 1) * 64;    // wave N offset
    const int m0   = blockIdx.y * GTM;
    const int n0   = blockIdx.x * GTN;
    const int nk   = K / GTK;

    f32x8 acc[4][4];
#pragma unroll
    for (int i = 0; i < 4; ++i)
#pragma unroll
        for (int j = 0; j < 4; ++j) acc[i][j] = (f32x8)0.f;

    // one row (32 bf16 = 64B) per thread per tile, for both A and Bt
    const __bf16* gA = A  + (size_t)(m0 + tid) * K;
    const __bf16* gB = Bt + (size_t)(n0 + tid) * K;

#ifdef HAVE_ASYNC_LDS
    auto stage = [&](int kt, int buf) {
        const __bf16* a = gA + kt * GTK;
        const __bf16* b = gB + kt * GTK;
        __bf16* la = &lA[buf][tid][0];
        __bf16* lb = &lB[buf][tid][0];
#pragma unroll
        for (int i = 0; i < 4; ++i) {
            async_copy_b128(a + 8 * i, la + 8 * i);
            async_copy_b128(b + 8 * i, lb + 8 * i);
        }
    };
    stage(0, 0);
#endif

    for (int kt = 0; kt < nk; ++kt) {
        int buf;
#ifdef HAVE_ASYNC_LDS
        buf = kt & 1;
        if (kt + 1 < nk) {                 // overlap next-tile DMA with this WMMA
            stage(kt + 1, (kt + 1) & 1);
            WAIT_ASYNC(8);                 // previous 8 done, next 8 in flight
        } else {
            WAIT_ASYNC(0);
        }
        __syncthreads();
#else
        buf = 0;
        __syncthreads();
        {
            const __bf16* a = gA + kt * GTK;
            const __bf16* b = gB + kt * GTK;
#pragma unroll
            for (int i = 0; i < 4; ++i) {
                *(bf16x8*)&lA[0][tid][8 * i] = *(const bf16x8*)(a + 8 * i);
                *(bf16x8*)&lB[0][tid][8 * i] = *(const bf16x8*)(b + 8 * i);
            }
        }
        __syncthreads();
#endif
        Frag16 af[4], bf[4];
        const int fm = lane & 15, fk8 = (lane >> 4) * 8, fk16 = (lane >> 4) * 16;
#pragma unroll
        for (int i = 0; i < 4; ++i) {
            const __bf16* p = &lA[buf][wr + 16 * i + fm][fk8];
            af[i].h[0] = *(const bf16x8*)p;
            af[i].h[1] = *(const bf16x8*)(p + 16);
        }
#pragma unroll
        for (int j = 0; j < 4; ++j) {
            const __bf16* p = &lB[buf][wc + 16 * j + fm][fk8];
            bf[j].h[0] = *(const bf16x8*)p;
            bf[j].h[1] = *(const bf16x8*)(p + 16);
        }
#pragma unroll
        for (int i = 0; i < 4; ++i)
#pragma unroll
            for (int j = 0; j < 4; ++j)
                acc[i][j] = __builtin_amdgcn_wmma_f32_16x16x32_bf16(
                    false, af[i].v, false, bf[j].v, (short)0, acc[i][j], false, false);
#ifdef HAVE_ASYNC_LDS
        __syncthreads();                   // all reads done before buffer reuse
#endif
    }

    const int cm = m0 + wr + ((lane >> 4) << 3);
    const int cn = n0 + wc + (lane & 15);
#pragma unroll
    for (int i = 0; i < 4; ++i)
#pragma unroll
        for (int j = 0; j < 4; ++j) {
            float bv = bias ? bias[cn + 16 * j] : 0.f;
#pragma unroll
            for (int r = 0; r < 8; ++r)
                C[(size_t)(cm + 16 * i + r) * N + cn + 16 * j] = acc[i][j][r] + bv;
        }
}

// ============================================================== RoPE + QKV split
// qkv f32 [B*S, QKVD] -> q bf16 [B,NH,S,HD], k bf16 [B,NKV,S,HD],
//                        v bf16 TRANSPOSED [B,NKV,HD,S]  (attention B-fragment ready)
__global__ __launch_bounds__(256)
void rope_split(const float* __restrict__ qkv, const int* __restrict__ pos,
                __bf16* __restrict__ qb, __bf16* __restrict__ kb,
                __bf16* __restrict__ vbT) {
    const int bs = blockIdx.x;
    const int b  = bs / SEQ, s = bs % SEQ;
    const float p = (float)pos[bs];
    const float* row = qkv + (size_t)bs * QKVD;

    for (int c = threadIdx.x; c < QKVD; c += blockDim.x) {
        int h = c >> 7;
        int d = c & 127;
        float o = row[c];
        if (h < NHEAD + NKVH && d < ROTD) {
            float inv = __powf(10000.f, -(float)(d >> 1) / 32.0f);
            float fr  = p * inv;
            float sn, cs;
            __sincosf(fr, &sn, &cs);
            float x1 = row[c & ~1];
            float x2 = row[c | 1];
            o = (d & 1) ? (x2 * cs + x1 * sn) : (x1 * cs - x2 * sn);
        }
        __bf16 ob = (__bf16)o;
        if (h < NHEAD) {
            qb[(((size_t)b * NHEAD + h) * SEQ + s) * HDIM + d] = ob;
        } else if (h < NHEAD + NKVH) {
            int kh = h - NHEAD;
            kb[(((size_t)b * NKVH + kh) * SEQ + s) * HDIM + d] = ob;
        } else {
            int vh = h - NHEAD - NKVH;
            vbT[(((size_t)b * NKVH + vh) * HDIM + d) * SEQ + s] = ob;
        }
    }
}

// ============================================================== flash attention
// grid (S/64, NH, B), 128 threads = 4 waves, each wave owns 16 query rows.
// Q (A-frag), K (B-frag), V-transposed (B-frag) all loaded DIRECTLY from global
// (row-major layouts match the ISA fragment shapes). LDS used only for the
// per-wave P round-trip (C-layout -> A-fragment).
__global__ __launch_bounds__(128)
void attn_flash(const __bf16* __restrict__ qb, const __bf16* __restrict__ kb,
                const __bf16* __restrict__ vbT, __bf16* __restrict__ ctx) {
    __shared__ __bf16 Ps[4][16][64 + 8];       // per-wave P scratch [m][key]

    const int tid = threadIdx.x, lane = tid & 31, w = tid >> 5;
    const int b = blockIdx.z, h = blockIdx.y, qblk = blockIdx.x;
    const int kvh = h / (NHEAD / NKVH);
    const int qr  = qblk * 64 + w * 16;

    const __bf16* qbase = qb  + ((size_t)b * NHEAD + h)   * SEQ * HDIM;
    const __bf16* kbase = kb  + ((size_t)b * NKVH  + kvh) * SEQ * HDIM;
    const __bf16* vtb   = vbT + ((size_t)b * NKVH  + kvh) * (size_t)HDIM * SEQ;

    // Q fragments: 4 x (16x32 bf16) in registers for the whole kernel
    Frag16 qf[4];
    {
        int fm = lane & 15, fk8 = (lane >> 4) * 8;
        const __bf16* qrow = qbase + (size_t)(qr + fm) * HDIM;
#pragma unroll
        for (int kc = 0; kc < 4; ++kc) {
            const __bf16* p = qrow + kc * 32 + fk8;
            qf[kc].h[0] = *(const bf16x8*)p;
            qf[kc].h[1] = *(const bf16x8*)(p + 16);
        }
    }

    f32x8 oacc[8];
#pragma unroll
    for (int j = 0; j < 8; ++j) oacc[j] = (f32x8)0.f;
    float rowm[8], rowl[8];
#pragma unroll
    for (int r = 0; r < 8; ++r) { rowm[r] = -3.0e38f; rowl[r] = 0.f; }

    const float scale = 0.08838834764831845f;  // 1/sqrt(128)

    for (int kt = 0; kt <= qblk; ++kt) {
        const int kb0 = kt * 64;

        if (kt + 1 <= qblk) {  // hint next K/V tiles toward the caches
            __builtin_prefetch(kbase + (size_t)(kb0 + 64 + lane * 2) * HDIM, 0, 0);
            __builtin_prefetch(vtb + (size_t)(lane * 4) * SEQ + kb0 + 64, 0, 0);
        }

        // ---- S = Q K^T (K B-fragments from global: row=key, contiguous d)
        f32x8 sacc[4];
#pragma unroll
        for (int j = 0; j < 4; ++j) sacc[j] = (f32x8)0.f;
        {
            int fn = lane & 15, fk8 = (lane >> 4) * 8;
#pragma unroll
            for (int kc = 0; kc < 4; ++kc)
#pragma unroll
                for (int j = 0; j < 4; ++j) {
                    const __bf16* p =
                        kbase + (size_t)(kb0 + 16 * j + fn) * HDIM + kc * 32 + fk8;
                    Frag16 bfr;
                    bfr.h[0] = *(const bf16x8*)p;
                    bfr.h[1] = *(const bf16x8*)(p + 16);
                    sacc[j] = __builtin_amdgcn_wmma_f32_16x16x32_bf16(
                        false, qf[kc].v, false, bfr.v, (short)0, sacc[j], false, false);
                }
        }

        // ---- scale + causal mask
        const int mbase = qr + ((lane >> 4) << 3);
        const int nbase = kb0 + (lane & 15);
#pragma unroll
        for (int j = 0; j < 4; ++j) {
            int kg = nbase + 16 * j;
#pragma unroll
            for (int r = 0; r < 8; ++r) {
                float sv = sacc[j][r] * scale;
                if (kg > mbase + r) sv = -3.0e38f;
                sacc[j][r] = sv;
            }
        }

        // ---- online softmax (row reductions across 16 lanes of a half-wave)
        float pscale[8];
#pragma unroll
        for (int r = 0; r < 8; ++r) {
            float v = fmaxf(fmaxf(sacc[0][r], sacc[1][r]),
                            fmaxf(sacc[2][r], sacc[3][r]));
#pragma unroll
            for (int m = 1; m < 16; m <<= 1) v = fmaxf(v, __shfl_xor(v, m));
            float mnew = fmaxf(rowm[r], v);
            float sc   = __expf(rowm[r] - mnew);
            rowm[r] = mnew;
            float rs = 0.f;
#pragma unroll
            for (int j = 0; j < 4; ++j) {
                float pv = __expf(sacc[j][r] - mnew);
                sacc[j][r] = pv;
                rs += pv;
            }
#pragma unroll
            for (int m = 1; m < 16; m <<= 1) rs += __shfl_xor(rs, m);
            rowl[r]   = rowl[r] * sc + rs;
            pscale[r] = sc;
        }
#pragma unroll
        for (int j = 0; j < 8; ++j)
#pragma unroll
            for (int r = 0; r < 8; ++r) oacc[j][r] *= pscale[r];

        // ---- P (C layout) -> per-wave LDS -> A fragments
        {
            int pm = ((lane >> 4) << 3);
            int pn = lane & 15;
#pragma unroll
            for (int j = 0; j < 4; ++j)
#pragma unroll
                for (int r = 0; r < 8; ++r)
                    Ps[w][pm + r][16 * j + pn] = (__bf16)sacc[j][r];
        }
        __syncthreads();   // uniform trip count across waves; orders Ps store->load

        Frag16 pf[2];
        {
            int fm = lane & 15, fk8 = (lane >> 4) * 8;
#pragma unroll
            for (int kp = 0; kp < 2; ++kp) {
                const __bf16* p = &Ps[w][fm][kp * 32 + fk8];
                pf[kp].h[0] = *(const bf16x8*)p;
                pf[kp].h[1] = *(const bf16x8*)(p + 16);
            }
        }
        // ---- O += P.V  (V B-fragments from global transposed V: row=d, contig keys)
        {
            int fn = lane & 15, fk8 = (lane >> 4) * 8;
#pragma unroll
            for (int kp = 0; kp < 2; ++kp)
#pragma unroll
                for (int j = 0; j < 8; ++j) {
                    const __bf16* p =
                        vtb + (size_t)(16 * j + fn) * SEQ + kb0 + kp * 32 + fk8;
                    Frag16 bfr;
                    bfr.h[0] = *(const bf16x8*)p;
                    bfr.h[1] = *(const bf16x8*)(p + 16);
                    oacc[j] = __builtin_amdgcn_wmma_f32_16x16x32_bf16(
                        false, pf[kp].v, false, bfr.v, (short)0, oacc[j], false, false);
                }
        }
    }

    // ---- normalize and store ctx [B*S, NH*HD] bf16
    {
        int pm = ((lane >> 4) << 3);
        int pn = lane & 15;
#pragma unroll
        for (int j = 0; j < 8; ++j)
#pragma unroll
            for (int r = 0; r < 8; ++r) {
                float o = oacc[j][r] / rowl[r];
                size_t rowi = (size_t)b * SEQ + (qr + pm + r);
                ctx[rowi * (NHEAD * HDIM) + (size_t)h * HDIM + 16 * j + pn] = (__bf16)o;
            }
    }
}

// ============================================================== launch
extern "C" void kernel_launch(void* const* d_in, const int* in_sizes, int n_in,
                              void* d_out, int out_size, void* d_ws, size_t ws_size,
                              hipStream_t stream) {
    (void)in_sizes; (void)n_in; (void)out_size; (void)ws_size;
    const float* hs   = (const float*)d_in[0];
    const int*   pos  = (const int*)d_in[1];
    const float* Wqkv = (const float*)d_in[2];
    const float* bqkv = (const float*)d_in[3];
    const float* Wo   = (const float*)d_in[4];
    float* out = (float*)d_out;

    char* ws = (char*)d_ws;
    size_t off = 0;
    auto carve = [&](size_t bytes) {
        void* p = ws + off;
        off += (bytes + 255) & ~(size_t)255;
        return p;
    };
    __bf16* hsb    = (__bf16*)carve((size_t)MROWS * HID * 2);   // reused as ctx
    __bf16* wqkvT  = (__bf16*)carve((size_t)QKVD * HID * 2);    // [N][K]
    __bf16* woT    = (__bf16*)carve((size_t)HID * HID * 2);     // [N][K]
    float*  qkvf   = (float*) carve((size_t)MROWS * QKVD * 4);
    __bf16* qbuf   = (__bf16*)carve((size_t)BATCH * NHEAD * SEQ * HDIM * 2);
    __bf16* kbuf   = (__bf16*)carve((size_t)BATCH * NKVH  * SEQ * HDIM * 2);
    __bf16* vbufT  = (__bf16*)carve((size_t)BATCH * NKVH  * HDIM * SEQ * 2);
    __bf16* ctxb   = hsb;

    // 1) activation convert + weight convert/transpose (one-time prep)
    cvt_f32_bf16<<<2048, 256, 0, stream>>>(hs, hsb, (size_t)MROWS * HID);
    cvt_transpose_bf16<<<dim3(QKVD / 64, HID / 64), 256, 0, stream>>>(
        Wqkv, wqkvT, HID, QKVD);
    cvt_transpose_bf16<<<dim3(HID / 64, HID / 64), 256, 0, stream>>>(
        Wo, woT, HID, HID);

    // 2) QKV projection
    gemm_bf16<<<dim3(QKVD / GTN, MROWS / GTM), 128, 0, stream>>>(
        hsb, wqkvT, bqkv, qkvf, MROWS, QKVD, HID);

    // 3) RoPE + split (V written transposed)
    rope_split<<<MROWS, 256, 0, stream>>>(qkvf, pos, qbuf, kbuf, vbufT);

    // 4) causal GQA flash attention
    attn_flash<<<dim3(SEQ / 64, NHEAD, BATCH), 128, 0, stream>>>(
        qbuf, kbuf, vbufT, ctxb);

    // 5) output projection
    gemm_bf16<<<dim3(HID / GTN, MROWS / GTM), 128, 0, stream>>>(
        ctxb, woT, nullptr, out, MROWS, HID, HID);
}
